// BitNetLinearLayer_63634235457580
// MI455X (gfx1250) — matrix operand order repaired and checked by reading, not compiled, not run on hardware
//
#include <hip/hip_runtime.h>

// ---------------------------------------------------------------------------
// BitNet b1.58 linear layer for MI455X (gfx1250, wave32, WMMA)
//   scale = 1e-4 + mean(|W|);  q = clip(round(W/scale)) in {-1,0,1}
//   out   = scale * X @ q^T          X: 8192x2048 f32, W: 8192x2048 f32
// Compute-bound (~275 GFLOP vs ~0.4 GB HBM traffic) -> bf16 WMMA path.
// Block tile 128(M) x 256(N) x 32(K), 8 waves, 64x64 per wave (4x4 WMMA tiles).
// ---------------------------------------------------------------------------

typedef __attribute__((ext_vector_type(16))) __bf16 v16bf;
typedef __attribute__((ext_vector_type(8)))  float  v8f;

#define IN_F    2048
#define OUT_F   8192
#define M_TOT   8192   // 4 * 2048 rows of X
#define BM      128
#define BN      256
#define BK      32
#define THREADS 256
#define NELEM   (OUT_F * (size_t)IN_F)   // 16,777,216 (same as X element count)

// float -> bf16 via native convert (RNE); pack two into a dword
__device__ __forceinline__ unsigned pack_bf16(float lo, float hi) {
    union { __bf16 b; unsigned short s; } a, b;
    a.b = (__bf16)lo;
    b.b = (__bf16)hi;
    return (unsigned)a.s | ((unsigned)b.s << 16);
}

// ---------------------------------------------------------------------------
// Pass 1: per-block partial sums of |W| (fixed order -> deterministic)
// 1024 blocks x 256 threads, each thread reduces 16 float4 = 64 floats
// ---------------------------------------------------------------------------
__global__ void absmean_partial(const float* __restrict__ w,
                                float* __restrict__ partial) {
    __shared__ float red[THREADS];
    const int t = threadIdx.x;
    const float4* w4 = (const float4*)w;
    const size_t b4 = (size_t)blockIdx.x * 4096;  // 4096 float4 per block
    float s = 0.f;
#pragma unroll
    for (int i = 0; i < 16; ++i) {
        float4 v = w4[b4 + (size_t)t + (size_t)i * THREADS];
        s += fabsf(v.x) + fabsf(v.y) + fabsf(v.z) + fabsf(v.w);
    }
    red[t] = s;
    __syncthreads();
    for (int off = THREADS / 2; off > 0; off >>= 1) {
        if (t < off) red[t] += red[t + off];
        __syncthreads();
    }
    if (t == 0) partial[blockIdx.x] = red[0];
}

// ---------------------------------------------------------------------------
// Pass 2: reduce 1024 partials -> scale = 1e-4 + sum / N
// ---------------------------------------------------------------------------
__global__ void absmean_final(const float* __restrict__ partial,
                              float* __restrict__ scale_out) {
    __shared__ float red[THREADS];
    const int t = threadIdx.x;
    red[t] = partial[t] + partial[t + 256] + partial[t + 512] + partial[t + 768];
    __syncthreads();
    for (int off = THREADS / 2; off > 0; off >>= 1) {
        if (t < off) red[t] += red[t + off];
        __syncthreads();
    }
    if (t == 0) scale_out[0] = 1e-4f + red[0] / (float)NELEM;
}

// ---------------------------------------------------------------------------
// Pass 3: quantize W -> ternary bf16 ({-1,0,1} exact in bf16), 4 elems/thread
// ---------------------------------------------------------------------------
__global__ void quantize_w(const float* __restrict__ w,
                           const float* __restrict__ scale_p,
                           unsigned short* __restrict__ q) {
    const float s = scale_p[0];
    const size_t i = ((size_t)blockIdx.x * THREADS + threadIdx.x) * 4;
    float4 v = *(const float4*)(w + i);
    float a0 = fminf(1.f, fmaxf(-1.f, rintf(v.x / s)));
    float a1 = fminf(1.f, fmaxf(-1.f, rintf(v.y / s)));
    float a2 = fminf(1.f, fmaxf(-1.f, rintf(v.z / s)));
    float a3 = fminf(1.f, fmaxf(-1.f, rintf(v.w / s)));
    uint2 o;
    o.x = pack_bf16(a0, a1);
    o.y = pack_bf16(a2, a3);
    *(uint2*)(q + i) = o;
}

// ---------------------------------------------------------------------------
// Pass 4: GEMM  out[m,n] = scale * sum_k X[m,k] * q[n,k]
// ---------------------------------------------------------------------------
__global__ void __launch_bounds__(THREADS)
bitnet_gemm(const float* __restrict__ x,
            const unsigned short* __restrict__ q,
            const float* __restrict__ scale_p,
            float* __restrict__ out) {
    __shared__ __align__(16) unsigned short As[2][BM][BK];  // 2 x 8 KB
    __shared__ __align__(16) unsigned short Bs[2][BN][BK];  // 2 x 16 KB

    const int t    = threadIdx.x;
    const int lane = t & 31;
    const int wave = t >> 5;
    const int wm   = wave >> 2;          // 0..1 -> 64-row strip of M
    const int wn   = wave & 3;           // 0..3 -> 64-col strip of N
    const int m0   = blockIdx.y * BM;
    const int n0   = blockIdx.x * BN;

    // Fragment gather coordinates (ISA 7.12.2, 16-bit A 16x32 / B 32x16)
    const int r  = lane & 15;            // row (A) / column (B) within tile
    const int kA = (lane >> 4) * 8;      // A: lanes 0-15 K=0-7&16-23; 16-31 K=8-15&24-31
    const int kB = (lane >> 4) * 8;      // B: lanes 0-15 K=0-7&16-23; 16-31 K=8-15&24-31

    v8f acc[4][4];
#pragma unroll
    for (int i = 0; i < 4; ++i)
#pragma unroll
        for (int j = 0; j < 4; ++j)
            acc[i][j] = (v8f){0.f, 0.f, 0.f, 0.f, 0.f, 0.f, 0.f, 0.f};

    // --- stage X tile (f32 -> bf16 via native cvt): 128x32 = 1024 float4 chunks
    auto stageA = [&](int buf, int kk) {
#pragma unroll
        for (int c = 0; c < 4; ++c) {
            int chunk = t + c * THREADS;       // 0..1023
            int row   = chunk >> 3;            // 8 float4 per row
            int k4    = (chunk & 7) * 4;
            float4 v  = *(const float4*)(x + (size_t)(m0 + row) * IN_F + kk + k4);
            uint2 p;
            p.x = pack_bf16(v.x, v.y);
            p.y = pack_bf16(v.z, v.w);
            *(uint2*)&As[buf][row][k4] = p;
        }
    };
    // --- stage q tile (bf16): 256x32 halfs = 1024 uint4 chunks
    auto stageB = [&](int buf, int kk) {
#pragma unroll
        for (int c = 0; c < 4; ++c) {
            int chunk = t + c * THREADS;       // 0..1023
            int row   = chunk >> 2;            // 4 uint4 per row
            int k8    = (chunk & 3) * 8;
            uint4 v   = *(const uint4*)(q + (size_t)(n0 + row) * IN_F + kk + k8);
            *(uint4*)&Bs[buf][row][k8] = v;
        }
    };

    stageA(0, 0);
    stageB(0, 0);
    __syncthreads();

    union Frag { uint4 u[2]; v16bf v; };

    const int KSTEPS = IN_F / BK;   // 64
    for (int kb = 0; kb < KSTEPS; ++kb) {
        const int cur = kb & 1;
        if (kb + 1 < KSTEPS) {      // prefetch next tile into other buffer
            stageA(cur ^ 1, (kb + 1) * BK);
            stageB(cur ^ 1, (kb + 1) * BK);
        }

        Frag a[4];
#pragma unroll
        for (int i = 0; i < 4; ++i) {
            const unsigned short* p = &As[cur][wm * 64 + i * 16 + r][0];
            a[i].u[0] = *(const uint4*)(p + kA);
            a[i].u[1] = *(const uint4*)(p + kA + 16);
        }
#pragma unroll
        for (int j = 0; j < 4; ++j) {
            Frag b;   // one B fragment live at a time -> bounded reg pressure
            const unsigned short* p = &Bs[cur][wn * 64 + j * 16 + r][0];
            b.u[0] = *(const uint4*)(p + kB);
            b.u[1] = *(const uint4*)(p + kB + 16);
#pragma unroll
            for (int i = 0; i < 4; ++i)
                acc[i][j] = __builtin_amdgcn_wmma_f32_16x16x32_bf16(
                    /*neg_a=*/false, a[i].v, /*neg_b=*/false, b.v,
                    /*c_mod=*/(short)0, acc[i][j],
                    /*reuse_a=*/false, /*reuse_b=*/false);
        }

        __syncthreads();
    }

    // Epilogue: C/D 16x16 f32 layout -> m = vgpr + 8*(lane>=16), n = lane%16
    const float scale = scale_p[0];
    const int cm = (lane >> 4) * 8;
    const int cn = lane & 15;
#pragma unroll
    for (int i = 0; i < 4; ++i)
#pragma unroll
        for (int j = 0; j < 4; ++j)
#pragma unroll
            for (int v = 0; v < 8; ++v) {
                const int mm = m0 + wm * 64 + i * 16 + cm + v;
                const int nn = n0 + wn * 64 + j * 16 + cn;
                out[(size_t)mm * OUT_F + nn] = scale * acc[i][j][v];
            }
}

// ---------------------------------------------------------------------------
extern "C" void kernel_launch(void* const* d_in, const int* in_sizes, int n_in,
                              void* d_out, int out_size, void* d_ws, size_t ws_size,
                              hipStream_t stream) {
    const float* x = (const float*)d_in[0];       // (4,2048,2048) f32
    const float* w = (const float*)d_in[1];       // (8192,2048)  f32
    float* out = (float*)d_out;                   // (4,2048,8192) f32

    char* ws = (char*)d_ws;
    float* scale_p          = (float*)ws;                   // 4 B
    float* partials         = (float*)(ws + 1024);          // 4 KB
    unsigned short* q_bf16  = (unsigned short*)(ws + 8192); // 32 MiB ternary bf16

    absmean_partial<<<1024, THREADS, 0, stream>>>(w, partials);
    absmean_final<<<1, THREADS, 0, stream>>>(partials, scale_p);
    quantize_w<<<(unsigned)(NELEM / (THREADS * 4)), THREADS, 0, stream>>>(w, scale_p, q_bf16);

    dim3 grid(OUT_F / BN, M_TOT / BM);  // 32 x 64 blocks
    bitnet_gemm<<<grid, THREADS, 0, stream>>>(x, q_bf16, scale_p, out);
}
